// PointRM_84731114816202
// MI455X (gfx1250) — matrix operand order repaired and checked by reading, not compiled, not run on hardware
//
#include <hip/hip_runtime.h>
#include <hip/hip_fp16.h>

typedef __attribute__((ext_vector_type(16))) _Float16 v16h;
typedef __attribute__((ext_vector_type(8)))  _Float16 v8h;
typedef __attribute__((ext_vector_type(8)))  float    v8f;

#define BATCH      8
#define NPTS       8192
#define NUM_GROUP  512
#define GROUP_SIZE 32
#define CCH        384
#define BN_EPS     1e-5f
#define STD_EPS    1e-5f
#define LDS_STRIDE 392   // 384 + 8 pad: conflict-free b128 reads (784B row = 196 dwords == 4 mod 64)

// ---------------------------------------------------------------- utilities
__global__ void zero_stats_kernel(float* stats) {
    int t = threadIdx.x;
    if (t < 24) stats[t] = 0.0f;   // 16 accum + 8 std slots
}

// ---------------------------------------------------------------- FPS
// One block per batch. dists held in registers (8 pts/thread @ 1024 threads).
__global__ __launch_bounds__(1024)
void fps_kernel(const float* __restrict__ xyz, float* __restrict__ center) {
    __shared__ float rv[1024];
    __shared__ int   ri[1024];
    __shared__ float curp[3];
    const int b   = blockIdx.x;
    const int tid = threadIdx.x;
    const float* P = xyz + (size_t)b * NPTS * 3;

    float px[8], py[8], pz[8], dist[8];
    #pragma unroll
    for (int j = 0; j < 8; ++j) {
        int p = tid + j * 1024;
        px[j] = P[p*3+0]; py[j] = P[p*3+1]; pz[j] = P[p*3+2];
        dist[j] = 3.4e38f;
    }
    int cur = 0;
    for (int it = 0; it < NUM_GROUP; ++it) {
        if (tid == 0) { curp[0]=P[cur*3]; curp[1]=P[cur*3+1]; curp[2]=P[cur*3+2]; }
        __syncthreads();
        float cx = curp[0], cy = curp[1], cz = curp[2];
        if (tid == 0) {   // scan emits 'last' before update: c_idx[0]=0
            float* c = center + ((size_t)b*NUM_GROUP + it)*3;
            c[0]=cx; c[1]=cy; c[2]=cz;
        }
        float bv = -1.0f; int bi = 0x7fffffff;
        #pragma unroll
        for (int j = 0; j < 8; ++j) {
            float dx = px[j]-cx, dy = py[j]-cy, dz = pz[j]-cz;
            float d  = dx*dx + dy*dy + dz*dz;
            dist[j]  = fminf(dist[j], d);
            int p = tid + j*1024;
            if (dist[j] > bv || (dist[j] == bv && p < bi)) { bv = dist[j]; bi = p; }
        }
        rv[tid] = bv; ri[tid] = bi;
        __syncthreads();
        for (int s2 = 512; s2 > 0; s2 >>= 1) {
            if (tid < s2) {
                float v = rv[tid+s2]; int i2 = ri[tid+s2];
                if (v > rv[tid] || (v == rv[tid] && i2 < ri[tid])) { rv[tid]=v; ri[tid]=i2; }
            }
            __syncthreads();
        }
        cur = ri[0];
        __syncthreads();
    }
}

// ---------------------------------------------------------------- kNN + diff + stats
// One block per (b,s). d2 in LDS, 32x iterative stable argmin (== stable top_k).
__global__ __launch_bounds__(256)
void knn_kernel(const float* __restrict__ xyz, const float* __restrict__ center,
                float* __restrict__ diffb, float* __restrict__ stats) {
    __shared__ float d2s[NPTS];
    __shared__ float rv[256];
    __shared__ int   ri[256];
    __shared__ int   sel[GROUP_SIZE];
    __shared__ float ssum, ssq;
    const int bs = blockIdx.x;
    const int b  = bs >> 9;
    const int tid = threadIdx.x;
    const float* P = xyz + (size_t)b * NPTS * 3;
    const float* c = center + (size_t)bs * 3;
    const float cx = c[0], cy = c[1], cz = c[2];

    for (int p = tid; p < NPTS; p += 256) {
        float dx = P[p*3]-cx, dy = P[p*3+1]-cy, dz = P[p*3+2]-cz;
        d2s[p] = dx*dx + dy*dy + dz*dz;
    }
    if (tid == 0) { ssum = 0.0f; ssq = 0.0f; }
    __syncthreads();

    for (int k = 0; k < GROUP_SIZE; ++k) {
        float bv = 3.4e38f; int bi = 0x7fffffff;
        for (int p = tid; p < NPTS; p += 256) {
            float v = d2s[p];
            if (v < bv || (v == bv && p < bi)) { bv = v; bi = p; }
        }
        rv[tid] = bv; ri[tid] = bi;
        __syncthreads();
        for (int s2 = 128; s2 > 0; s2 >>= 1) {
            if (tid < s2) {
                float v = rv[tid+s2]; int i2 = ri[tid+s2];
                if (v < rv[tid] || (v == rv[tid] && i2 < ri[tid])) { rv[tid]=v; ri[tid]=i2; }
            }
            __syncthreads();
        }
        if (tid == 0) { sel[k] = ri[0]; d2s[ri[0]] = 3.4e38f; }
        __syncthreads();
    }
    if (tid < GROUP_SIZE) {
        int idx = sel[tid];
        float dx = P[idx*3]-cx, dy = P[idx*3+1]-cy, dz = P[idx*3+2]-cz;
        float* d = diffb + ((size_t)bs*GROUP_SIZE + tid)*3;
        d[0]=dx; d[1]=dy; d[2]=dz;
        atomicAdd(&ssum, dx + dy + dz);
        atomicAdd(&ssq,  dx*dx + dy*dy + dz*dz);
    }
    __syncthreads();
    if (tid == 0) {
        atomicAdd(&stats[b*2+0], ssum);
        atomicAdd(&stats[b*2+1], ssq);
    }
}

__global__ void std_kernel(const float* __restrict__ stats, float* __restrict__ stdv) {
    int b = threadIdx.x;
    if (b < BATCH) {
        const float n = (float)(NUM_GROUP * GROUP_SIZE * 3);
        float su = stats[b*2], sq = stats[b*2+1];
        float var = (sq - su*su/n) / (n - 1.0f);   // ddof=1
        stdv[b] = sqrtf(fmaxf(var, 0.0f));
    }
}

// ---------------------------------------------------------------- embed: W(384x6) conv + BN + ReLU -> f16
__global__ __launch_bounds__(256)
void embed_kernel(const float* __restrict__ diffb, const float* __restrict__ center,
                  const float* __restrict__ stdv,  const float* __restrict__ alpha,
                  const float* __restrict__ beta,
                  const float* __restrict__ Wt, const float* __restrict__ bt,
                  const float* __restrict__ gt, const float* __restrict__ bbt,
                  __half* __restrict__ h0) {
    __shared__ float x6[6][GROUP_SIZE];
    const int m = blockIdx.x;
    const int b = m >> 9;
    const int tid = threadIdx.x;
    if (tid < GROUP_SIZE) {
        float inv = 1.0f / (stdv[b] + STD_EPS);
        const float* d = diffb + ((size_t)m*GROUP_SIZE + tid)*3;
        const float* c = center + (size_t)m*3;
        #pragma unroll
        for (int ch = 0; ch < 3; ++ch) {
            x6[ch][tid]   = alpha[ch] * (d[ch] * inv) + beta[ch];
            x6[3+ch][tid] = c[ch];
        }
    }
    __syncthreads();
    const float rs = rsqrtf(1.0f + BN_EPS);
    for (int p = tid; p < CCH*GROUP_SIZE; p += 256) {
        int o = p >> 5, l = p & 31;
        float acc = bt[o];
        #pragma unroll
        for (int ch = 0; ch < 6; ++ch) acc += Wt[o*6+ch] * x6[ch][l];
        float y = acc * (gt[o]*rs) + bbt[o];
        h0[(size_t)m*CCH*GROUP_SIZE + p] = __float2half(fmaxf(y, 0.0f));
    }
}

// ---------------------------------------------------------------- weight f32 -> f16
__global__ void wcvt_kernel(const float* __restrict__ W0, const float* __restrict__ W1,
                            const float* __restrict__ W2, const float* __restrict__ W3,
                            __half* __restrict__ out) {
    const int n = CCH*CCH;
    for (int p = blockIdx.x*blockDim.x + threadIdx.x; p < 4*n; p += gridDim.x*blockDim.x) {
        int w = p / n, i = p - w*n;
        const float* src = (w==0) ? W0 : (w==1) ? W1 : (w==2) ? W2 : W3;
        out[p] = __float2half(src[i]);
    }
}

// ---------------------------------------------------------------- WMMA conv: Out = relu(bn(W@X + bias) [+ resid])
// X layout: [img][c][L] f16, one block handles 32 consecutive columns of one image.
// Block = 256 thr = 8 waves; wave -> 3 o-tiles of 16; accumulators: cols 0-15 / 16-31.
__global__ __launch_bounds__(256)
void conv_wmma_kernel(const __half* __restrict__ X, const __half* __restrict__ Wh,
                      const float* __restrict__ bias, const float* __restrict__ gamma,
                      const float* __restrict__ bbeta, const __half* __restrict__ resid,
                      __half* __restrict__ Out, int strideC, int chunksPerImage) {
    __shared__ _Float16 Xt[GROUP_SIZE * LDS_STRIDE];   // transposed [l][c], padded
    const int blk   = blockIdx.x;
    const int img   = blk / chunksPerImage;
    const int chunk = blk - img * chunksPerImage;
    const size_t base = (size_t)img * CCH * strideC + (size_t)chunk * 32;
    const int tid = threadIdx.x;

    // stage chunk transposed into LDS: b128 global loads along contiguous l,
    // 8x ds_store_b16 scatter (16B-aligned: base, c*strideC, g all multiples of 8 halfs)
    for (int p = tid; p < CCH*4; p += 256) {
        int cc = p >> 2, g = (p & 3) * 8;
        v8h v = *(const v8h*)(X + base + (size_t)cc*strideC + g);
        #pragma unroll
        for (int j = 0; j < 8; ++j) Xt[(g + j)*LDS_STRIDE + cc] = v[j];
    }
    __syncthreads();

    const int lane = tid & 31;
    const int wave = tid >> 5;
    const int n    = lane & 15;
    const int hl   = lane >> 4;
    const float rs = rsqrtf(1.0f + BN_EPS);

    for (int t = 0; t < 3; ++t) {
        const int o0 = (wave*3 + t) * 16;
        v8f acc0 = {0,0,0,0,0,0,0,0};
        v8f acc1 = {0,0,0,0,0,0,0,0};
        const _Float16* wrow = (const _Float16*)Wh + (size_t)(o0 + n) * CCH;
        for (int k0 = 0; k0 < CCH; k0 += 32) {
            // A fragment (16x32 f16): lane half hl -> K = hl*16 .. hl*16+15 (contiguous)
            v8h alo = *(const v8h*)(wrow + k0 + hl*16);
            v8h ahi = *(const v8h*)(wrow + k0 + hl*16 + 8);
            v16h a  = __builtin_shufflevector(alo, ahi, 0,1,2,3,4,5,6,7,8,9,10,11,12,13,14,15);
            // B fragments (32x16 f16): col n(+16), K = k0 + hl*16 + e, contiguous in Xt row
            const _Float16* b0p = &Xt[n*LDS_STRIDE + k0 + hl*16];
            v8h b0lo = *(const v8h*)(b0p);
            v8h b0hi = *(const v8h*)(b0p + 8);
            v16h bf0 = __builtin_shufflevector(b0lo, b0hi, 0,1,2,3,4,5,6,7,8,9,10,11,12,13,14,15);
            const _Float16* b1p = &Xt[(n+16)*LDS_STRIDE + k0 + hl*16];
            v8h b1lo = *(const v8h*)(b1p);
            v8h b1hi = *(const v8h*)(b1p + 8);
            v16h bf1 = __builtin_shufflevector(b1lo, b1hi, 0,1,2,3,4,5,6,7,8,9,10,11,12,13,14,15);

            acc0 = __builtin_amdgcn_wmma_f32_16x16x32_f16(false, a, false, bf0, (short)0, acc0, false, false);
            acc1 = __builtin_amdgcn_wmma_f32_16x16x32_f16(false, a, false, bf1, (short)0, acc1, false, false);
        }
        // fused epilogue: bias + BN (+ residual) + ReLU, f16 store.
        // Per-lane rows are o0+hl*8 .. +7 (32B-aligned) -> vector param loads.
        const int rbase = o0 + hl*8;
        v8f gv  = *(const v8f*)(gamma + rbase);
        v8f bvv = *(const v8f*)(bias  + rbase);
        v8f bbv = *(const v8f*)(bbeta + rbase);
        #pragma unroll
        for (int r = 0; r < 8; ++r) {
            const int row = rbase + r;
            const float scale = gv[r] * rs;
            const size_t a0 = base + (size_t)row * strideC + n;
            const size_t a1 = a0 + 16;
            float y0 = (acc0[r] + bvv[r]) * scale + bbv[r];
            float y1 = (acc1[r] + bvv[r]) * scale + bbv[r];
            if (resid) { y0 += __half2float(resid[a0]); y1 += __half2float(resid[a1]); }
            Out[a0] = __float2half(fmaxf(y0, 0.0f));
            Out[a1] = __float2half(fmaxf(y1, 0.0f));
        }
    }
}

// ---------------------------------------------------------------- max-pool over K
__global__ void pool_kernel(const __half* __restrict__ in, __half* __restrict__ feat) {
    const int total = BATCH * CCH * NUM_GROUP;
    for (int p = blockIdx.x*blockDim.x + threadIdx.x; p < total; p += gridDim.x*blockDim.x) {
        int b = p / (CCH*NUM_GROUP);
        int rem = p - b*(CCH*NUM_GROUP);
        int cc = rem / NUM_GROUP;
        int s  = rem - cc*NUM_GROUP;
        const __half* src = in + (((size_t)b*NUM_GROUP + s)*CCH + cc)*GROUP_SIZE;
        float mx = -3.4e38f;
        #pragma unroll
        for (int l = 0; l < GROUP_SIZE; ++l) mx = fmaxf(mx, __half2float(src[l]));
        feat[p] = __float2half(mx);   // (b, c, s) layout == p
    }
}

// ---------------------------------------------------------------- final transpose (b,c,s)->(b,s,c) f32
__global__ void out_kernel(const __half* __restrict__ t4, float* __restrict__ out) {
    const int total = BATCH * NUM_GROUP * CCH;
    for (int p = blockIdx.x*blockDim.x + threadIdx.x; p < total; p += gridDim.x*blockDim.x) {
        int b = p / (NUM_GROUP*CCH);
        int rem = p - b*(NUM_GROUP*CCH);
        int s = rem / CCH;
        int cc = rem - s*CCH;
        out[p] = __half2float(t4[((size_t)b*CCH + cc)*NUM_GROUP + s]);
    }
}

// ---------------------------------------------------------------- launch
extern "C" void kernel_launch(void* const* d_in, const int* in_sizes, int n_in,
                              void* d_out, int out_size, void* d_ws, size_t ws_size,
                              hipStream_t stream) {
    (void)in_sizes; (void)n_in; (void)out_size; (void)ws_size;
    const float* xyz   = (const float*)d_in[0];
    const float* alpha = (const float*)d_in[1];
    const float* beta  = (const float*)d_in[2];
    const float* Wt    = (const float*)d_in[3];
    const float* bt    = (const float*)d_in[4];
    const float* gt    = (const float*)d_in[5];
    const float* bbt   = (const float*)d_in[6];
    const float* Wpre1 = (const float*)d_in[7];
    const float* bpre1 = (const float*)d_in[8];
    const float* gpre1 = (const float*)d_in[9];
    const float* bbpre1= (const float*)d_in[10];
    const float* Wpre2 = (const float*)d_in[11];
    const float* bpre2 = (const float*)d_in[12];
    const float* gpre2 = (const float*)d_in[13];
    const float* bbpre2= (const float*)d_in[14];
    const float* Wpos1 = (const float*)d_in[15];
    const float* bpos1 = (const float*)d_in[16];
    const float* gpos1 = (const float*)d_in[17];
    const float* bbpos1= (const float*)d_in[18];
    const float* Wpos2 = (const float*)d_in[19];
    const float* bpos2 = (const float*)d_in[20];
    const float* gpos2 = (const float*)d_in[21];
    const float* bbpos2= (const float*)d_in[22];

    char* ws = (char*)d_ws;
    size_t off = 0;
    auto alloc = [&](size_t bytes) -> void* {
        void* p = ws + off;
        off = (off + bytes + 255) & ~(size_t)255;
        return p;
    };
    float*  center = (float*)alloc((size_t)BATCH*NUM_GROUP*3*sizeof(float));
    float*  diffb  = (float*)alloc((size_t)BATCH*NUM_GROUP*GROUP_SIZE*3*sizeof(float));
    float*  stats  = (float*)alloc(24*sizeof(float));   // [16 accum][8 std]
    float*  stdv   = stats + 16;
    __half* Wf16   = (__half*)alloc((size_t)4*CCH*CCH*sizeof(__half));
    __half* bufA   = (__half*)alloc((size_t)BATCH*NUM_GROUP*CCH*GROUP_SIZE*sizeof(__half));
    __half* bufB   = (__half*)alloc((size_t)BATCH*NUM_GROUP*CCH*GROUP_SIZE*sizeof(__half));
    __half* bufC   = (__half*)alloc((size_t)BATCH*CCH*NUM_GROUP*sizeof(__half));
    __half* bufD   = (__half*)alloc((size_t)BATCH*CCH*NUM_GROUP*sizeof(__half));

    const int MPOS = BATCH * NUM_GROUP;   // 4096

    zero_stats_kernel<<<1, 32, 0, stream>>>(stats);
    fps_kernel<<<BATCH, 1024, 0, stream>>>(xyz, center);
    knn_kernel<<<MPOS, 256, 0, stream>>>(xyz, center, diffb, stats);
    std_kernel<<<1, 32, 0, stream>>>(stats, stdv);
    wcvt_kernel<<<1024, 256, 0, stream>>>(Wpre1, Wpre2, Wpos1, Wpos2, Wf16);
    embed_kernel<<<MPOS, 256, 0, stream>>>(diffb, center, stdv, alpha, beta,
                                           Wt, bt, gt, bbt, bufA);
    // pre res-block (strideC = L = 32, 1 chunk/image, 4096 images)
    conv_wmma_kernel<<<MPOS, 256, 0, stream>>>(bufA, Wf16 + 0*CCH*CCH,
                                               bpre1, gpre1, bbpre1, nullptr,
                                               bufB, GROUP_SIZE, 1);
    conv_wmma_kernel<<<MPOS, 256, 0, stream>>>(bufB, Wf16 + 1*CCH*CCH,
                                               bpre2, gpre2, bbpre2, bufA,
                                               bufB, GROUP_SIZE, 1);  // in-place per-chunk safe
    pool_kernel<<<2048, 256, 0, stream>>>(bufB, bufC);
    // pos res-block (strideC = L = 512, 16 chunks/image, 8 images)
    conv_wmma_kernel<<<BATCH*16, 256, 0, stream>>>(bufC, Wf16 + 2*CCH*CCH,
                                                   bpos1, gpos1, bbpos1, nullptr,
                                                   bufD, NUM_GROUP, 16);
    conv_wmma_kernel<<<BATCH*16, 256, 0, stream>>>(bufD, Wf16 + 3*CCH*CCH,
                                                   bpos2, gpos2, bbpos2, bufC,
                                                   bufD, NUM_GROUP, 16);
    out_kernel<<<2048, 256, 0, stream>>>(bufD, (float*)d_out);
}
